// Model_4166118277238
// MI455X (gfx1250) — matrix operand order repaired and checked by reading, not compiled, not run on hardware
//
#include <hip/hip_runtime.h>
#include <hip/hip_bf16.h>

typedef __attribute__((ext_vector_type(16))) _Float16 v16h;
typedef __attribute__((ext_vector_type(8)))  _Float16 v8h;
typedef __attribute__((ext_vector_type(8)))  float    v8f;

// ---------------------------------------------------------------------------
// Model dims
// ---------------------------------------------------------------------------
#define NB     512
#define GTOT   32901          // sum(GATE_SIZES)
#define EDIM   69
#define DDIM   128
#define CH     64
#define FC1IN  28224          // 64*21*21
#define FC1OUT 384
#define KS_FC1 7              // K-split factor for fc1 (882 = 7*126)

__device__ __forceinline__ float sigmoidf_(float x) { return 1.0f / (1.0f + expf(-x)); }

// ---------------------------------------------------------------------------
// f32 -> f16 convert
// ---------------------------------------------------------------------------
__global__ void k_cvt16(const float* __restrict__ in, _Float16* __restrict__ out, int n) {
    int i = blockIdx.x * blockDim.x + threadIdx.x;
    if (i < n) out[i] = (_Float16)in[i];
}

// NCHW f32 (3 ch) -> NHWC f16
__global__ void k_x_to_nhwc(const float* __restrict__ in, _Float16* __restrict__ out) {
    const int HW = 84 * 84;
    int i = blockIdx.x * blockDim.x + threadIdx.x;   // over NB*HW*3, c minor
    if (i >= NB * HW * 3) return;
    int c = i % 3, t = i / 3;
    int p = t % HW, n = t / HW;
    out[i] = (_Float16)in[((long)n * 3 + c) * HW + p];
}

// ---------------------------------------------------------------------------
// Pack weights into WMMA B-fragment layout, NT rounded up to a multiple of 4
// and zero-filled so the GEMM inner loop needs no guards.
// B fragment (16x16x32 f16): lane holds column n = lane%16; element e holds
// K = 8*(lane/16) + (e<8 ? e : e+8).  Storage: [kt][nt][lane][16] halves, contiguous.
// mode 0: W is [Nout x K] row-major -> B[k][n]=W[n*K+k]
// mode 1: W is [K x Nout] row-major -> B[k][n]=W[k*Nout+n]
// ---------------------------------------------------------------------------
__global__ void k_pack_w(const float* __restrict__ W, _Float16* __restrict__ out,
                         int K, int Nout, int KT, int NTpad, int mode) {
    int i = blockIdx.x * blockDim.x + threadIdx.x;
    int total = KT * NTpad * 32 * 16;
    if (i >= total) return;
    int e    = i & 15;
    int lane = (i >> 4) & 31;
    int nt   = (i >> 9) % NTpad;
    int kt   = i / (512 * NTpad);
    int k = kt * 32 + ((lane < 16) ? 0 : 8) + ((e < 8) ? e : e + 8);
    int n = nt * 16 + (lane & 15);
    float v = 0.0f;
    if (k < K && n < Nout) v = (mode == 0) ? W[(long)n * K + k] : W[(long)k * Nout + n];
    out[i] = (_Float16)v;
}

// ---------------------------------------------------------------------------
// Implicit-GEMM 5x5 pad-2 conv, NHWC f16 in -> NHWC f16 out (bias+ReLU fused).
// Compile-time dims => all index math is mul/shift.  Each wave computes
// 32 output pixels x 64 channels (2 M-tiles, B fragment reused twice).
// Launch is exact (MT divisible by waves/block): no guards, straight line.
// ---------------------------------------------------------------------------
template <int CIN, int H, int W, int K, int KT>
__global__ void k_conv_wmma(const _Float16* __restrict__ xh,
                            const _Float16* __restrict__ wp,
                            const float* __restrict__ bias,
                            _Float16* __restrict__ out) {
    constexpr int HW = H * W;
    int wid  = threadIdx.x >> 5;
    int lane = threadIdx.x & 31;
    int mtile = blockIdx.x * (blockDim.x >> 5) + wid;

    int nimg[2], oy[2], ox[2];
#pragma unroll
    for (int mi = 0; mi < 2; ++mi) {
        int p = mtile * 32 + mi * 16 + (lane & 15);
        nimg[mi] = p / HW;                        // compile-time HW -> mulhi
        int rem = p % HW;
        oy[mi] = rem / W;
        ox[mi] = rem % W;
    }

    v8f acc[2][4] = {};
    for (int kt = 0; kt < KT; ++kt) {
        int kbase = kt * 32;
        // speculative prefetch of next k-step's B fragments (OOB is dropped)
        __builtin_prefetch(wp + (((long)(kt + 1) * 4) * 32 + lane) * 16, 0, 3);
        v16h a[2];
#pragma unroll
        for (int mi = 0; mi < 2; ++mi) {
            const _Float16* xbase = xh + (long)nimg[mi] * HW * CIN;
#pragma unroll
            for (int e = 0; e < 16; ++e) {
                int kl = (lane < 16) ? ((e < 8) ? e : e + 8)
                                     : ((e < 8) ? e + 8 : e + 16);
                int k  = kbase + kl;
                int c  = k / 25, r5 = k % 25;     // compile-time consts
                int rr = r5 / 5, ss = r5 % 5;
                int iy = oy[mi] + rr - 2, ix = ox[mi] + ss - 2;
                bool ok = (k < K) && (iy >= 0) && (iy < H) && (ix >= 0) && (ix < W);
                int idx = ok ? ((iy * W + ix) * CIN + c) : 0;   // clamped, valid
                _Float16 v = xbase[idx];
                a[mi][e] = ok ? v : (_Float16)0;
            }
        }
#pragma unroll
        for (int nt = 0; nt < 4; ++nt) {
            v16h b = *(const v16h*)(wp + (((long)kt * 4 + nt) * 32 + lane) * 16);
            acc[0][nt] = __builtin_amdgcn_wmma_f32_16x16x32_f16(
                false, a[0], false, b, (short)0, acc[0][nt], false, false);
            acc[1][nt] = __builtin_amdgcn_wmma_f32_16x16x32_f16(
                false, a[1], false, b, (short)0, acc[1][nt], false, false);
        }
    }
    // D layout: reg r, lane L -> M = r + 8*(L/16), N = L%16.
    // NHWC out: row m (pixel) * 64 + channel -> contiguous stores across lanes.
    int ch = lane & 15;
#pragma unroll
    for (int mi = 0; mi < 2; ++mi) {
#pragma unroll
        for (int nt = 0; nt < 4; ++nt) {
            float bv = bias[nt * 16 + ch];
#pragma unroll
            for (int r = 0; r < 8; ++r) {
                int m = mtile * 32 + mi * 16 + r + ((lane >> 4) << 3);
                float v = acc[mi][nt][r] + bv;
                v = v > 0.0f ? v : 0.0f;
                out[(long)m * CH + nt * 16 + ch] = (_Float16)v;
            }
        }
    }
}

// ---------------------------------------------------------------------------
// WMMA GEMM: out[MxNout] = A[MxKpad](f16,row-major) * Bpacked (+bias,relu).
// NTpad is a multiple of 4 (padded pack) => no guards around WMMA.
// Exact launch: waves == mtiles*NG, divisible by 8.
// ---------------------------------------------------------------------------
__global__ void k_gemm_wmma(const _Float16* __restrict__ A,
                            const _Float16* __restrict__ wp,
                            const float* __restrict__ bias,
                            float* __restrict__ out,
                            int Kpad, int Nout, int KT, int NTpad, int NG,
                            int relu) {
    int wid  = threadIdx.x >> 5;
    int lane = threadIdx.x & 31;
    int gw = blockIdx.x * (blockDim.x >> 5) + wid;
    int mtile = gw / NG;
    int ng    = gw % NG;

    int row = mtile * 16 + (lane & 15);
    const _Float16* arow = A + (long)row * Kpad;
    int aoff = (lane < 16) ? 0 : 8;

    v8f acc[4] = {};
    for (int kt = 0; kt < KT; ++kt) {
        int off = kt * 32 + aoff;
        __builtin_prefetch(arow + off + 32, 0, 3);
        v8h lo = *(const v8h*)(arow + off);
        v8h hi = *(const v8h*)(arow + off + 16);
        v16h a = __builtin_shufflevector(lo, hi, 0, 1, 2, 3, 4, 5, 6, 7,
                                         8, 9, 10, 11, 12, 13, 14, 15);
#pragma unroll
        for (int nt = 0; nt < 4; ++nt) {
            int ntile = ng * 4 + nt;
            v16h b = *(const v16h*)(wp + (((long)kt * NTpad + ntile) * 32 + lane) * 16);
            acc[nt] = __builtin_amdgcn_wmma_f32_16x16x32_f16(
                false, a, false, b, (short)0, acc[nt], false, false);
        }
    }
#pragma unroll
    for (int nt = 0; nt < 4; ++nt) {
        int ntile = ng * 4 + nt;
#pragma unroll
        for (int r = 0; r < 8; ++r) {
            int m = mtile * 16 + r + ((lane >> 4) << 3);
            int n = ntile * 16 + (lane & 15);
            if (n < Nout) {
                float v = acc[nt][r] + (bias ? bias[n] : 0.0f);
                if (relu) v = v > 0.0f ? v : 0.0f;
                out[(long)m * Nout + n] = v;
            }
        }
    }
}

// ---------------------------------------------------------------------------
// K-split WMMA GEMM (for fc1): each wave handles KTn k-tiles starting at
// ks*KTn, writing f32 partials part[ks][m][n].  waves = mtiles*NG*KS.
// ---------------------------------------------------------------------------
__global__ void k_gemm_wmma_ksplit(const _Float16* __restrict__ A,
                                   const _Float16* __restrict__ wp,
                                   float* __restrict__ part,
                                   int Kpad, int Nout, int KTn, int NTpad,
                                   int NG, int KS) {
    int wid  = threadIdx.x >> 5;
    int lane = threadIdx.x & 31;
    int gw = blockIdx.x * (blockDim.x >> 5) + wid;
    int ks   = gw % KS;
    int rest = gw / KS;
    int ng    = rest % NG;
    int mtile = rest / NG;

    int row = mtile * 16 + (lane & 15);
    const _Float16* arow = A + (long)row * Kpad;
    int aoff = (lane < 16) ? 0 : 8;
    int kt0 = ks * KTn;

    v8f acc[4] = {};
    for (int kt = kt0; kt < kt0 + KTn; ++kt) {
        int off = kt * 32 + aoff;
        __builtin_prefetch(arow + off + 32, 0, 3);
        v8h lo = *(const v8h*)(arow + off);
        v8h hi = *(const v8h*)(arow + off + 16);
        v16h a = __builtin_shufflevector(lo, hi, 0, 1, 2, 3, 4, 5, 6, 7,
                                         8, 9, 10, 11, 12, 13, 14, 15);
#pragma unroll
        for (int nt = 0; nt < 4; ++nt) {
            int ntile = ng * 4 + nt;
            v16h b = *(const v16h*)(wp + (((long)kt * NTpad + ntile) * 32 + lane) * 16);
            acc[nt] = __builtin_amdgcn_wmma_f32_16x16x32_f16(
                false, a, false, b, (short)0, acc[nt], false, false);
        }
    }
    float* pbase = part + (long)ks * NB * Nout;
#pragma unroll
    for (int nt = 0; nt < 4; ++nt) {
        int ntile = ng * 4 + nt;
#pragma unroll
        for (int r = 0; r < 8; ++r) {
            int m = mtile * 16 + r + ((lane >> 4) << 3);
            int n = ntile * 16 + (lane & 15);
            pbase[(long)m * Nout + n] = acc[nt][r];
        }
    }
}

// reduce K-split partials + bias + ReLU, write f16 (fc1 activation)
__global__ void k_fc1_finish(const float* __restrict__ part, const float* __restrict__ bias,
                             _Float16* __restrict__ outh) {
    int i = blockIdx.x * blockDim.x + threadIdx.x;  // over NB*FC1OUT
    if (i >= NB * FC1OUT) return;
    int n = i % FC1OUT;
    float s = bias[n];
#pragma unroll
    for (int ks = 0; ks < KS_FC1; ++ks) s += part[(long)ks * NB * FC1OUT + i];
    s = s > 0.0f ? s : 0.0f;
    outh[i] = (_Float16)s;
}

// ---------------------------------------------------------------------------
// 3x3 stride-2 pad-1 maxpool, NHWC f16 -> NHWC f16 (64 channels, square maps)
// ---------------------------------------------------------------------------
template <int H, int OH>
__global__ void k_maxpool_nhwc(const _Float16* __restrict__ in, _Float16* __restrict__ out) {
    int i = blockIdx.x * blockDim.x + threadIdx.x;  // over NB*OH*OH*64, c minor
    if (i >= NB * OH * OH * CH) return;
    int c = i & 63, t = i >> 6;
    int ox = t % OH, t2 = t / OH;
    int oy = t2 % OH, n = t2 / OH;
    const _Float16* base = in + (long)n * H * H * CH + c;
    float m = -3.4e38f;
#pragma unroll
    for (int dy = 0; dy < 3; ++dy)
#pragma unroll
        for (int dx = 0; dx < 3; ++dx) {
            int iy = oy * 2 - 1 + dy, ix = ox * 2 - 1 + dx;
            if (iy >= 0 && iy < H && ix >= 0 && ix < H) {
                float v = (float)base[(iy * H + ix) * CH];
                m = v > m ? v : m;
            }
        }
    out[i] = (_Float16)m;
}

// ---------------------------------------------------------------------------
// LRN over channels (window c-2..c+1), NHWC f16 -> NHWC f16, 64 channels.
// ---------------------------------------------------------------------------
__global__ void k_lrn_nhwc(const _Float16* __restrict__ in, _Float16* __restrict__ out,
                           int total) {
    int i = blockIdx.x * blockDim.x + threadIdx.x;
    if (i >= total) return;
    int c = i & 63;
    const _Float16* base = in + (i - c);
    float win = 0.0f;
#pragma unroll
    for (int d = -2; d <= 1; ++d) {
        int cc = c + d;
        if (cc >= 0 && cc < CH) {
            float v = (float)base[cc];
            win += v * v;
        }
    }
    win *= 0.25f;
    float x = (float)base[c];
    float denom = powf(1.0f + (0.001f / 9.0f) * win, 0.75f);
    out[i] = (_Float16)(x / denom);
}

// ---------------------------------------------------------------------------
// maxpool 42->21 reading NHWC, writing NCHW-flattened f16 rows for fc1
// ---------------------------------------------------------------------------
__global__ void k_pool_to_flat(const _Float16* __restrict__ in, _Float16* __restrict__ out) {
    int i = blockIdx.x * blockDim.x + threadIdx.x;  // over NB*28224
    if (i >= NB * FC1IN) return;
    int within = i % FC1IN, n = i / FC1IN;
    int c = within / 441;
    int r = within % 441;
    int oy = r / 21, ox = r % 21;
    const _Float16* base = in + (long)n * 42 * 42 * CH + c;
    float m = -3.4e38f;
#pragma unroll
    for (int dy = 0; dy < 3; ++dy)
#pragma unroll
        for (int dx = 0; dx < 3; ++dx) {
            int iy = oy * 2 - 1 + dy, ix = ox * 2 - 1 + dx;
            if (iy >= 0 && iy < 42 && ix >= 0 && ix < 42) {
                float v = (float)base[(iy * 42 + ix) * CH];
                m = v > m ? v : m;
            }
        }
    out[i] = (_Float16)m;
}

// ---------------------------------------------------------------------------
// task_emb = [relu(fc2 out), one_hot(y)] ; also copied to d_out segment
// ---------------------------------------------------------------------------
__global__ void k_task_emb(const float* __restrict__ fc2o, const int* __restrict__ y,
                           float* __restrict__ temb, float* __restrict__ out_temb) {
    int i = blockIdx.x * blockDim.x + threadIdx.x;
    if (i >= NB * EDIM) return;
    int e = i % EDIM, n = i / EDIM;
    float v = (e < CH) ? fc2o[n * CH + e] : ((y[n] == (e - CH)) ? 1.0f : 0.0f);
    temb[i] = v;
    out_temb[i] = v;
}

// ---------------------------------------------------------------------------
// Sequential GRU encoder scan (512 steps), 1 block / 384 threads
// ---------------------------------------------------------------------------
__global__ void k_enc_gru(const float* __restrict__ temb,
                          const float* __restrict__ wih, const float* __restrict__ whh,
                          const float* __restrict__ bih, const float* __restrict__ bhh,
                          float* __restrict__ enc_states) {
    __shared__ float h[DDIM];
    __shared__ float xt[EDIM];
    __shared__ float gi[3 * DDIM];
    __shared__ float gh[3 * DDIM];
    int tid = threadIdx.x;                     // 384 threads
    if (tid < DDIM) h[tid] = 0.0f;
    __syncthreads();
    float bi = bih[tid], bh = bhh[tid];
    const float* wi = wih + (long)tid * EDIM;
    const float* wh = whh + (long)tid * DDIM;
    for (int t = 0; t < NB; ++t) {
        if (tid < EDIM) xt[tid] = temb[t * EDIM + tid];
        __syncthreads();
        float a = bi;
        for (int k = 0; k < EDIM; ++k) a += xt[k] * wi[k];
        float b = bh;
        for (int k = 0; k < DDIM; ++k) b += h[k] * wh[k];
        gi[tid] = a;
        gh[tid] = b;
        __syncthreads();
        if (tid < DDIM) {
            float r  = sigmoidf_(gi[tid] + gh[tid]);
            float z  = sigmoidf_(gi[DDIM + tid] + gh[DDIM + tid]);
            float nn = tanhf(gi[2 * DDIM + tid] + r * gh[2 * DDIM + tid]);
            float hn = (1.0f - z) * nn + z * h[tid];
            h[tid] = hn;
            enc_states[t * DDIM + tid] = hn;
        }
        __syncthreads();
    }
}

__global__ void k_mean_hidden(const float* __restrict__ enc_states, float* __restrict__ hidden) {
    int j = threadIdx.x;                      // 128 threads
    float s = 0.0f;
    for (int t = 0; t < NB; ++t) s += enc_states[t * DDIM + j];
    hidden[j] = s * (1.0f / (float)NB);
}

// ---------------------------------------------------------------------------
// Decoder GRU combine (x=0 so gi=dec_bih); gh has dec_bhh folded in via GEMM
// bias.  Writes f32 (for reference shape) is unnecessary: only the final GEMM
// consumes it, so write f16 directly.
// ---------------------------------------------------------------------------
__global__ void k_dec_combine(const float* __restrict__ gh, const float* __restrict__ enc_states,
                              const float* __restrict__ bih, _Float16* __restrict__ dechh) {
    int i = blockIdx.x * blockDim.x + threadIdx.x;
    if (i >= NB * DDIM) return;
    int j = i % DDIM, n = i / DDIM;
    const float* g = gh + (long)n * 3 * DDIM;
    float r  = sigmoidf_(bih[j] + g[j]);
    float z  = sigmoidf_(bih[DDIM + j] + g[DDIM + j]);
    float nn = tanhf(bih[2 * DDIM + j] + r * g[2 * DDIM + j]);
    dechh[i] = (_Float16)((1.0f - z) * nn + z * enc_states[i]);
}

// recon = dec_out[::-1]
__global__ void k_recon_rev(const float* __restrict__ deco, float* __restrict__ out) {
    int i = blockIdx.x * blockDim.x + threadIdx.x;
    if (i >= NB * EDIM) return;
    int e = i % EDIM, n = i / EDIM;
    out[n * EDIM + e] = deco[(NB - 1 - n) * EDIM + e];
}

// ---------------------------------------------------------------------------
// Cluster layers: softmax over the singleton axis gives probs==1, and every
// layer reads `hidden` (not emb), so only cl2 (C=1) determines emb.
// ---------------------------------------------------------------------------
__global__ void k_cluster(const float* __restrict__ hidden,
                          const float* __restrict__ cl2w, const float* __restrict__ cl2b,
                          float* __restrict__ cat) {
    __shared__ float hs[DDIM];
    int e = threadIdx.x;                      // 128 threads
    hs[e] = hidden[e];
    __syncthreads();
    float a = cl2b[e];
    const float* w = cl2w + (long)e * DDIM;
    for (int d = 0; d < DDIM; ++d) a += hs[d] * w[d];
    cat[e] = hs[e];
    cat[DDIM + e] = tanhf(a);
}

__global__ void k_gate(const float* __restrict__ cat, const float* __restrict__ gw,
                       const float* __restrict__ gb, float* __restrict__ out) {
    __shared__ float cs[2 * DDIM];
    int tid = threadIdx.x;
    if (tid < 2 * DDIM) cs[tid] = cat[tid];
    __syncthreads();
    int i = blockIdx.x * blockDim.x + tid;
    if (i >= GTOT) return;
    float a = gb[i];
    for (int k = 0; k < 2 * DDIM; ++k) a += cs[k] * gw[(long)k * GTOT + i];
    out[i] = sigmoidf_(a);
}

// ---------------------------------------------------------------------------
// Host: orchestrate the whole forward pass on `stream`
// ---------------------------------------------------------------------------
extern "C" void kernel_launch(void* const* d_in, const int* in_sizes, int n_in,
                              void* d_out, int out_size, void* d_ws, size_t ws_size,
                              hipStream_t stream) {
    (void)in_sizes; (void)n_in; (void)out_size; (void)ws_size;

    const float* x    = (const float*)d_in[0];
    const int*   y    = (const int*)  d_in[1];
    const float* c1w  = (const float*)d_in[2];
    const float* c1b  = (const float*)d_in[3];
    const float* c2w  = (const float*)d_in[4];
    const float* c2b  = (const float*)d_in[5];
    const float* fc1w = (const float*)d_in[6];
    const float* fc1b = (const float*)d_in[7];
    const float* fc2w = (const float*)d_in[8];
    const float* fc2b = (const float*)d_in[9];
    const float* ewih = (const float*)d_in[10];
    const float* ewhh = (const float*)d_in[11];
    const float* ebih = (const float*)d_in[12];
    const float* ebhh = (const float*)d_in[13];
    const float* dwhh = (const float*)d_in[15];
    const float* dbih = (const float*)d_in[16];
    const float* dbhh = (const float*)d_in[17];
    const float* dwt  = (const float*)d_in[18];
    const float* dbias= (const float*)d_in[19];
    const float* cl2w = (const float*)d_in[26];
    const float* cl2b = (const float*)d_in[27];
    const float* gw   = (const float*)d_in[29];
    const float* gb   = (const float*)d_in[30];
    float* out = (float*)d_out;

    // ---- workspace carving ----
    size_t off = 0;
    auto carve = [&](size_t bytes) -> char* {
        off = (off + 255) & ~(size_t)255;
        char* p = (char*)d_ws + off;
        off += bytes;
        return p;
    };
    const long HW1 = 84L * 84;                // 7056
    const long HW2 = 42L * 42;                // 1764
    _Float16* xh    = (_Float16*)carve((long)NB * HW1 * 3 * 2);   // NHWC
    _Float16* wp1   = (_Float16*)carve(3 * 4 * 512 * 2);          // KT=3 NT=4
    _Float16* c1h   = (_Float16*)carve((long)NB * HW1 * CH * 2);  // NHWC
    _Float16* p1a   = (_Float16*)carve((long)NB * HW2 * CH * 2);  // pooled NHWC
    _Float16* p1h   = (_Float16*)carve((long)NB * HW2 * CH * 2);  // LRN'd NHWC
    _Float16* wp2   = (_Float16*)carve(50 * 4 * 512 * 2);         // KT=50 NT=4
    _Float16* c2h   = (_Float16*)carve((long)NB * HW2 * CH * 2);  // NHWC
    _Float16* l2h   = (_Float16*)carve((long)NB * HW2 * CH * 2);  // NHWC
    _Float16* flath = (_Float16*)carve((long)NB * FC1IN * 2);     // NCHW flat
    _Float16* wpf1  = (_Float16*)carve((long)882 * 24 * 512 * 2); // KT=882 NTpad=24
    float*    part1 = (float*)   carve((long)KS_FC1 * NB * FC1OUT * 4);
    _Float16* fc1h  = (_Float16*)carve((long)NB * FC1OUT * 2);
    _Float16* wpf2  = (_Float16*)carve(12 * 4 * 512 * 2);         // KT=12 NTpad=4
    float*    fc2o  = (float*)   carve((long)NB * CH * 4);
    float*    temb  = (float*)   carve((long)NB * EDIM * 4);
    float*    encs  = (float*)   carve((long)NB * DDIM * 4);
    _Float16* ench  = (_Float16*)carve((long)NB * DDIM * 2);
    _Float16* wpdh  = (_Float16*)carve(4 * 24 * 512 * 2);         // KT=4 NTpad=24
    float*    ghb   = (float*)   carve((long)NB * 3 * DDIM * 4);
    _Float16* dechh = (_Float16*)carve((long)NB * DDIM * 2);
    _Float16* wpdw  = (_Float16*)carve(4 * 8 * 512 * 2);          // KT=4 NTpad=8
    float*    deco  = (float*)   carve((long)NB * EDIM * 4);
    float*    hid   = (float*)   carve(DDIM * 4);
    float*    catb  = (float*)   carve(2 * DDIM * 4);

    auto blks = [](long n, int b) { return (unsigned)((n + b - 1) / b); };

    // ---- stage 1: conv1 -> relu -> maxpool -> LRN ----
    k_x_to_nhwc<<<blks((long)NB * HW1 * 3, 256), 256, 0, stream>>>(x, xh);
    k_pack_w<<<blks(3 * 4 * 512, 256), 256, 0, stream>>>(c1w, wp1, 75, CH, 3, 4, 0);
    // MT = NB*HW1/32 = 112896 waves, /8 = 14112 blocks (exact)
    k_conv_wmma<3, 84, 84, 75, 3><<<14112, 256, 0, stream>>>(xh, wp1, c1b, c1h);
    k_maxpool_nhwc<84, 42><<<blks((long)NB * HW2 * CH, 256), 256, 0, stream>>>(c1h, p1a);
    k_lrn_nhwc<<<blks((long)NB * HW2 * CH, 256), 256, 0, stream>>>(p1a, p1h, (int)(NB * HW2 * CH));

    // ---- stage 2: conv2 -> relu -> LRN -> maxpool(->NCHW flat) ----
    k_pack_w<<<blks(50 * 4 * 512, 256), 256, 0, stream>>>(c2w, wp2, 1600, CH, 50, 4, 0);
    // MT = NB*HW2/32 = 28224 waves, /8 = 3528 blocks (exact)
    k_conv_wmma<64, 42, 42, 1600, 50><<<3528, 256, 0, stream>>>(p1h, wp2, c2b, c2h);
    k_lrn_nhwc<<<blks((long)NB * HW2 * CH, 256), 256, 0, stream>>>(c2h, l2h, (int)(NB * HW2 * CH));
    k_pool_to_flat<<<blks((long)NB * FC1IN, 256), 256, 0, stream>>>(l2h, flath);

    // ---- fc1 (K-split WMMA + fused reduce/bias/relu/f16) -> fc2 -> task_emb ----
    k_pack_w<<<blks((long)882 * 24 * 512, 256), 256, 0, stream>>>(fc1w, wpf1, FC1IN, FC1OUT, 882, 24, 0);
    // waves = 32 mtiles * 6 NG * 7 KS = 1344, /8 = 168 blocks (exact)
    k_gemm_wmma_ksplit<<<168, 256, 0, stream>>>(flath, wpf1, part1, FC1IN, FC1OUT,
                                                882 / KS_FC1, 24, 6, KS_FC1);
    k_fc1_finish<<<blks((long)NB * FC1OUT, 256), 256, 0, stream>>>(part1, fc1b, fc1h);
    k_pack_w<<<blks(12 * 4 * 512, 256), 256, 0, stream>>>(fc2w, wpf2, FC1OUT, CH, 12, 4, 0);
    // waves = 32*1 = 32, /8 = 4 blocks (exact)
    k_gemm_wmma<<<4, 256, 0, stream>>>(fc1h, wpf2, fc2b, fc2o, FC1OUT, CH, 12, 4, 1, 1);
    k_task_emb<<<blks((long)NB * EDIM, 256), 256, 0, stream>>>(fc2o, y, temb, out + NB * EDIM);

    // ---- encoder scan (sequential) + mean hidden ----
    k_enc_gru<<<1, 384, 0, stream>>>(temb, ewih, ewhh, ebih, ebhh, encs);
    k_mean_hidden<<<1, DDIM, 0, stream>>>(encs, hid);

    // ---- decoder GRU (batch) + output GEMM + reverse ----
    k_cvt16<<<blks((long)NB * DDIM, 256), 256, 0, stream>>>(encs, ench, NB * DDIM);
    k_pack_w<<<blks(4 * 24 * 512, 256), 256, 0, stream>>>(dwhh, wpdh, DDIM, 3 * DDIM, 4, 24, 0);
    // waves = 32*6 = 192, /8 = 24 blocks (exact)
    k_gemm_wmma<<<24, 256, 0, stream>>>(ench, wpdh, dbhh, ghb, DDIM, 3 * DDIM, 4, 24, 6, 0);
    k_dec_combine<<<blks((long)NB * DDIM, 256), 256, 0, stream>>>(ghb, encs, dbih, dechh);
    k_pack_w<<<blks(4 * 8 * 512, 256), 256, 0, stream>>>(dwt, wpdw, DDIM, EDIM, 4, 8, 1);
    // waves = 32*2 = 64, /8 = 8 blocks (exact)
    k_gemm_wmma<<<8, 256, 0, stream>>>(dechh, wpdw, dbias, deco, DDIM, EDIM, 4, 8, 2, 0);
    k_recon_rev<<<blks((long)NB * EDIM, 256), 256, 0, stream>>>(deco, out);

    // ---- clusters (only cl2 is live) + gates ----
    k_cluster<<<1, DDIM, 0, stream>>>(hid, cl2w, cl2b, catb);
    k_gate<<<blks(GTOT, 256), 256, 0, stream>>>(catb, gw, gb, out + 2 * NB * EDIM);
}
